// PredictModel_8057358647606
// MI455X (gfx1250) — matrix operand order repaired and checked by reading, not compiled, not run on hardware
//
#include <hip/hip_runtime.h>
#include <hip/hip_bf16.h>

// ---------------------------------------------------------------------------
// GNN link prediction pipeline for MI455X (gfx1250, wave32, WMMA).
// N=6144 nodes, F=H=128 features, E=16384 edges, top_k=32.
// Heavy GEMMs: v_wmma_f32_16x16x32_f16; GCN aggregation GEMMs stage A tiles
// in LDS via async global->LDS copies (ASYNCcnt) with double buffering.
// ---------------------------------------------------------------------------

#define NND   6144
#define FD    128
#define HD    128
#define ED    16384
#define TOPK  32
#define KC    256     // K-chunk staged in LDS for the aggregation GEMM

typedef __attribute__((ext_vector_type(16))) _Float16 v16h;
typedef __attribute__((ext_vector_type(8)))  _Float16 v8h;
typedef __attribute__((ext_vector_type(8)))  float    v8f;
typedef int v4i __attribute__((vector_size(16)));   // matches builtin param type

#define AS1 __attribute__((address_space(1)))
#define AS3 __attribute__((address_space(3)))

#if __has_builtin(__builtin_amdgcn_global_load_async_to_lds_b128) && \
    __has_builtin(__builtin_amdgcn_s_wait_asynccnt)
#define HAVE_ASYNC 1
#endif

#ifdef HAVE_ASYNC
#define ASYNC_COPY16(gp, lp)                                        \
  __builtin_amdgcn_global_load_async_to_lds_b128(                   \
      (AS1 v4i*)(gp), (AS3 v4i*)(lp), 0, 0)
#define WAIT_ASYNC(n) __builtin_amdgcn_s_wait_asynccnt(n)
#else
#define ASYNC_COPY16(gp, lp) (*(v8h*)(lp) = *(const v8h*)(gp))
#define WAIT_ASYNC(n) ((void)0)
#endif

// ---------------------------------------------------------------------------
// WMMA fragment helpers (ISA 7.12.2 layouts, wave32)
// A 16x32 f16: lane = hlf*16 + m ; elems 0..7 -> K = k0+8*hlf+0..7,
//                                  elems 8..15 -> K = k0+16+8*hlf+0..7
// B 32x16 f16: lane = hlf*16 + n ; elems 0..7 -> K = k0+8*hlf+0..7,
//                                  elems 8..15 -> K = k0+16+8*hlf+0..7
// C/D 16x16 f32: lane = hlf*16 + n ; vgpr r -> M = r + 8*hlf
// ---------------------------------------------------------------------------

__device__ __forceinline__ v16h frag_from_2x8(v8h lo, v8h hi) {
  v16h a;
#pragma unroll
  for (int i = 0; i < 8; ++i) { a[i] = lo[i]; a[i + 8] = hi[i]; }
  return a;
}

// A row-major [M,K]: elems at k0+8h+{0..7} and k0+16+8h+{0..7}
__device__ __forceinline__ v16h load_a_frag(const _Float16* __restrict__ A,
                                            int lda, int row, int k0, int hlf) {
  const _Float16* p = A + (size_t)row * lda + k0 + 8 * hlf;
  return frag_from_2x8(*(const v8h*)(p), *(const v8h*)(p + 16));
}

// B supplied transposed (row-major [Ncols, K]) -> computes A @ B^T
__device__ __forceinline__ v16h load_bt_frag(const _Float16* __restrict__ Bt,
                                             int ldb, int k0, int col, int hlf) {
  const _Float16* p = Bt + (size_t)col * ldb + k0 + 8 * hlf;
  return frag_from_2x8(*(const v8h*)(p), *(const v8h*)(p + 16));
}

// ---------------------------------------------------------------------------
// Generic wave-tiled WMMA GEMM: each wave computes one 16x16 tile of
//   C = [ELU]( A[M,K] @ Bt[Nc,K]^T (+ bias) )
// Optional f32 output Cf (stride ldc) and/or f16 output Ch (stride ldch;
// TRANSCH stores Ch transposed as [Nc, M] for feeding later B^T operands).
// blockDim = 256 (8 waves), one 16x16 tile per wave.
// ---------------------------------------------------------------------------
template <bool DOELU, bool DOBIAS, bool TRANSCH>
__global__ void wmma_gemm(const _Float16* __restrict__ A, int lda,
                          const _Float16* __restrict__ Bt, int ldb,
                          const float* __restrict__ bias,
                          float* __restrict__ Cf, int ldc,
                          _Float16* __restrict__ Ch, int ldch,
                          int tilesN, int Kdim) {
  int wave = blockIdx.x * 8 + (threadIdx.x >> 5);
  int lane = threadIdx.x & 31;
  int tm = wave / tilesN;
  int tn = wave - tm * tilesN;
  int hlf = lane >> 4;
  int mn  = lane & 15;
  int row = tm * 16 + mn;
  int col = tn * 16 + mn;

  v8f acc = {};
  for (int k0 = 0; k0 < Kdim; k0 += 32) {
    if (k0 + 128 < Kdim)
      __builtin_prefetch(A + (size_t)row * lda + k0 + 128, 0, 1);
    v16h a = load_a_frag(A, lda, row, k0, hlf);
    v16h b = load_bt_frag(Bt, ldb, k0, col, hlf);
    acc = __builtin_amdgcn_wmma_f32_16x16x32_f16(
        false, a, false, b, (short)0, acc, false, false);
  }

  float bv = DOBIAS ? bias[col] : 0.0f;
#pragma unroll
  for (int r = 0; r < 8; ++r) {
    int m = tm * 16 + r + 8 * hlf;
    float v = acc[r] + bv;
    if (DOELU) v = (v > 0.0f) ? v : expm1f(v);
    if (Cf) Cf[(size_t)m * ldc + col] = v;
    if (Ch) {
      if (TRANSCH) Ch[(size_t)col * ldch + m] = (_Float16)v;
      else         Ch[(size_t)m * ldch + col] = (_Float16)v;
    }
  }
}

// ---------------------------------------------------------------------------
// GCN aggregation GEMM: C[N,128] = ELU( Anorm[N,N] @ Bt[128,N]^T ).
// One workgroup per 16-row strip; 8 waves cover the 8 column tiles and share
// the A strip, which is streamed through LDS in double-buffered KC-chunks
// using async global->LDS copies (tracked by ASYNCcnt).
// ---------------------------------------------------------------------------
__global__ void gcn_agg(const _Float16* __restrict__ A,
                        const _Float16* __restrict__ Bt,
                        float* __restrict__ Cf, int ldc,
                        _Float16* __restrict__ Ch, int ldch, int n) {
  __shared__ _Float16 abuf[2][16 * KC];
  int tid  = threadIdx.x;
  int wave = tid >> 5, lane = tid & 31;
  int row0 = blockIdx.x * 16;
  int hlf  = lane >> 4, mn = lane & 15;
  int col  = wave * 16 + mn;

  auto issue = [&](int buf, int k0) {
    // 16 rows x KC halfs = 512 x 16B quanta; each thread moves 2 quanta
#pragma unroll
    for (int qi = 0; qi < 2; ++qi) {
      int q    = tid + qi * 256;
      int r    = q >> 5;            // row within strip
      int koff = (q & 31) * 8;      // 8 halfs = 16B per quantum
      const _Float16* gp = A + (size_t)(row0 + r) * n + k0 + koff;
      _Float16*       lp = &abuf[buf][r * KC + koff];
      ASYNC_COPY16(gp, lp);
    }
  };

  issue(0, 0);
  v8f acc = {};
  const int nch = n / KC;
  for (int c = 0; c < nch; ++c) {
    if (c + 1 < nch) {
      issue((c + 1) & 1, (c + 1) * KC);
      WAIT_ASYNC(2);      // chunk c landed; chunk c+1 still in flight
    } else {
      WAIT_ASYNC(0);
    }
    __syncthreads();
    const _Float16* ab = abuf[c & 1];
    const _Float16* pb0 = Bt + (size_t)col * n + c * KC;
#pragma unroll
    for (int kk = 0; kk < KC; kk += 32) {
      const _Float16* pa = ab + mn * KC + kk + 8 * hlf;
      v16h a = frag_from_2x8(*(const v8h*)pa, *(const v8h*)(pa + 16));
      const _Float16* pb = pb0 + kk + 8 * hlf;
      v16h b = frag_from_2x8(*(const v8h*)pb, *(const v8h*)(pb + 16));
      acc = __builtin_amdgcn_wmma_f32_16x16x32_f16(
          false, a, false, b, (short)0, acc, false, false);
    }
    __syncthreads();    // buffer (c&1) free for reuse at iteration c+2
  }

#pragma unroll
  for (int r = 0; r < 8; ++r) {
    int m = row0 + r + 8 * hlf;
    float v = acc[r];
    v = (v > 0.0f) ? v : expm1f(v);
    if (Cf) Cf[(size_t)m * ldc + col] = v;
    if (Ch) Ch[(size_t)m * ldch + col] = (_Float16)v;
  }
}

// ---------------------------------------------------------------------------
// Degree helpers for GCN normalization: a = adj + I, d1 = rowsum, d2 = colsum
// ---------------------------------------------------------------------------
__global__ void rowsum_rsqrt(const float* __restrict__ adj, float* __restrict__ out, int n) {
  __shared__ float red[256];
  int row = blockIdx.x, tid = threadIdx.x;
  const float* p = adj + (size_t)row * n;
  float s = 0.0f;
  for (int j = tid; j < n; j += 256) s += p[j];
  red[tid] = s;
  __syncthreads();
  for (int st = 128; st > 0; st >>= 1) {
    if (tid < st) red[tid] += red[tid + st];
    __syncthreads();
  }
  if (tid == 0) out[row] = rsqrtf(red[0] + 1.0f);   // +1 for identity
}

__global__ void colsum_rsqrt(const float* __restrict__ adj, float* __restrict__ out, int n) {
  int col = blockIdx.x * 256 + threadIdx.x;
  if (col >= n) return;
  float s = 0.0f;
  for (int i = 0; i < n; ++i) s += adj[(size_t)i * n + col];
  out[col] = rsqrtf(s + 1.0f);
}

__global__ void make_anorm(const float* __restrict__ adj,
                           const float* __restrict__ d1r,
                           const float* __restrict__ d2r,
                           _Float16* __restrict__ out, int n) {
  int idx = blockIdx.x * 256 + threadIdx.x;
  int i = idx / n, j = idx - i * n;
  float v = adj[idx] + (i == j ? 1.0f : 0.0f);
  out[idx] = (_Float16)(v * d1r[i] * d2r[j]);
}

// x -> f16 copy, row-normalized f16 copy, and left half of SAGE concat buffer
__global__ void prep_x(const float* __restrict__ x,
                       _Float16* __restrict__ x_h,
                       _Float16* __restrict__ xn_h,
                       _Float16* __restrict__ cat1) {
  __shared__ float red[128];
  int row = blockIdx.x, c = threadIdx.x;   // 128 threads
  float v = x[(size_t)row * FD + c];
  red[c] = v * v;
  __syncthreads();
  for (int st = 64; st > 0; st >>= 1) {
    if (c < st) red[c] += red[c + st];
    __syncthreads();
  }
  float inv = rsqrtf(red[0]);
  x_h [(size_t)row * FD + c]       = (_Float16)v;
  xn_h[(size_t)row * FD + c]       = (_Float16)(v * inv);
  cat1[(size_t)row * (2 * HD) + c] = (_Float16)v;
}

// transpose + f16: src row-major [rows, cols] -> dst row-major [cols, rows]
__global__ void f32_to_f16_t(const float* __restrict__ s, _Float16* __restrict__ d,
                             int rows, int cols) {
  int i = blockIdx.x * 256 + threadIdx.x;
  if (i < rows * cols) {
    int r = i / cols, c = i - r * cols;
    d[(size_t)c * rows + r] = (_Float16)s[i];
  }
}

// ---------------------------------------------------------------------------
// Per-row top-k (k=32) over the sim matrix via 32 block-wide argmax passes.
// ---------------------------------------------------------------------------
__global__ void topk_kernel(float* __restrict__ sim, int* __restrict__ ids,
                            float* __restrict__ vals, float* __restrict__ deginv, int n) {
  __shared__ float sv[256];
  __shared__ int   si[256];
  int row = blockIdx.x, tid = threadIdx.x;
  float* srow = sim + (size_t)row * n;
  if (tid == 0) srow[row] = -3.0e38f;   // mask self (reference: -NEG)
  __syncthreads();
  float degsum = 0.0f;
  for (int t = 0; t < TOPK; ++t) {
    float bm = -3.4e38f; int bi = n;
    for (int j = tid; j < n; j += 256) {
      float v = srow[j];
      if (v > bm || (v == bm && j < bi)) { bm = v; bi = j; }
    }
    sv[tid] = bm; si[tid] = bi;
    __syncthreads();
    for (int st = 128; st > 0; st >>= 1) {
      if (tid < st) {
        if (sv[tid + st] > sv[tid] ||
            (sv[tid + st] == sv[tid] && si[tid + st] < si[tid])) {
          sv[tid] = sv[tid + st]; si[tid] = si[tid + st];
        }
      }
      __syncthreads();
    }
    if (tid == 0) {
      ids [row * TOPK + t] = si[0];
      vals[row * TOPK + t] = sv[0];
      degsum += sv[0];
      srow[si[0]] = -3.4e38f;           // knock out for next pass
    }
    __syncthreads();
  }
  if (tid == 0) deginv[row] = 1.0f / fmaxf(degsum, 1e-5f);
}

// agg[row,:] = deginv[row] * sum_t vals[row,t] * Hin[ids[row,t],:]  (f16 out)
__global__ void agg_kernel(const float* __restrict__ Hin,
                           const int* __restrict__ ids,
                           const float* __restrict__ vals,
                           const float* __restrict__ deginv,
                           _Float16* __restrict__ out, int ldout, int off) {
  int row = blockIdx.x, c = threadIdx.x;   // 128 threads
  float acc = 0.0f;
#pragma unroll 4
  for (int t = 0; t < TOPK; ++t)
    acc += vals[row * TOPK + t] * Hin[(size_t)ids[row * TOPK + t] * HD + c];
  out[(size_t)row * ldout + off + c] = (_Float16)(acc * deginv[row]);
}

// ---------------------------------------------------------------------------
// Edge scoring: logit_e = src_h . (PredW @ dst_h) + PredB ; stable BCE term
// ---------------------------------------------------------------------------
__global__ void edge_kernel(const int* __restrict__ src, const int* __restrict__ dst,
                            const float* __restrict__ labels,
                            const float* __restrict__ h,
                            const float* __restrict__ PredW,
                            const float* __restrict__ PredB,
                            float* __restrict__ out,      // d_out: [1+e] = logit
                            float* __restrict__ terms) {
  __shared__ float red[256];
  int e = blockIdx.x, f = threadIdx.x;     // 256 threads
  const float* sh = h + (size_t)src[e] * 256;
  const float* dh = h + (size_t)dst[e] * 256;
  float t = 0.0f;
  for (int d = 0; d < 256; ++d)            // PredW reads coalesced over f
    t += sh[d] * PredW[d * 256 + f];
  red[f] = t * dh[f];
  __syncthreads();
  for (int st = 128; st > 0; st >>= 1) {
    if (f < st) red[f] += red[f + st];
    __syncthreads();
  }
  if (f == 0) {
    float l = red[0] + PredB[0];
    out[1 + e] = l;
    float y = labels[e];
    terms[e] = fmaxf(l, 0.0f) - l * y + log1pf(expf(-fabsf(l)));
  }
}

__global__ void loss_kernel(const float* __restrict__ terms, float* __restrict__ out) {
  __shared__ float red[256];
  int tid = threadIdx.x;
  float s = 0.0f;
  for (int i = tid; i < ED; i += 256) s += terms[i];
  red[tid] = s;
  __syncthreads();
  for (int st = 128; st > 0; st >>= 1) {
    if (tid < st) red[tid] += red[tid + st];
    __syncthreads();
  }
  if (tid == 0) out[0] = red[0] / (float)ED;
}

// ---------------------------------------------------------------------------
// Host launcher
// ---------------------------------------------------------------------------
static inline size_t al256(size_t x) { return (x + 255) & ~(size_t)255; }

extern "C" void kernel_launch(void* const* d_in, const int* in_sizes, int n_in,
                              void* d_out, int out_size, void* d_ws, size_t ws_size,
                              hipStream_t stream) {
  const int   N = NND;
  const int*   src    = (const int*)  d_in[0];
  const int*   dst    = (const int*)  d_in[1];
  const float* labels = (const float*)d_in[2];
  const float* adj    = (const float*)d_in[3];
  const float* x      = (const float*)d_in[4];
  const float* Wg1    = (const float*)d_in[5];
  const float* Wg2    = (const float*)d_in[6];
  const float* Wp1    = (const float*)d_in[7];
  const float* b1     = (const float*)d_in[8];
  const float* Ws1    = (const float*)d_in[9];
  const float* Wp2    = (const float*)d_in[10];
  const float* b2     = (const float*)d_in[11];
  const float* Ws2    = (const float*)d_in[12];
  const float* PredW  = (const float*)d_in[13];
  const float* PredB  = (const float*)d_in[14];
  float* out = (float*)d_out;

  // ---- workspace carve-up ----
  char* base = (char*)d_ws;
  size_t off = 0;
  auto alloc = [&](size_t bytes) { void* p = base + off; off = al256(off + bytes); return p; };

  float*     sim     = (float*)    alloc((size_t)N * N * 4);
  _Float16*  anorm   = (_Float16*) alloc((size_t)N * N * 2);
  _Float16*  x_h     = (_Float16*) alloc((size_t)N * FD * 2);
  _Float16*  xn_h    = (_Float16*) alloc((size_t)N * FD * 2);
  _Float16*  tT      = (_Float16*) alloc((size_t)HD * N * 2);  // transposed [128, N]
  _Float16*  z1h     = (_Float16*) alloc((size_t)N * HD * 2);
  _Float16*  cat1    = (_Float16*) alloc((size_t)N * 2 * HD * 2);
  _Float16*  cat2    = (_Float16*) alloc((size_t)N * 2 * HD * 2);
  float*     hp      = (float*)    alloc((size_t)N * HD * 4);
  float*     hbuf    = (float*)    alloc((size_t)N * 2 * HD * 4);
  float*     d1r     = (float*)    alloc((size_t)N * 4);
  float*     d2r     = (float*)    alloc((size_t)N * 4);
  float*     deginv  = (float*)    alloc((size_t)N * 4);
  int*       ids     = (int*)      alloc((size_t)N * TOPK * 4);
  float*     vals    = (float*)    alloc((size_t)N * TOPK * 4);
  _Float16*  Wg1t    = (_Float16*) alloc(FD * HD * 2);
  _Float16*  Wg2t    = (_Float16*) alloc(HD * HD * 2);
  _Float16*  Wp1t    = (_Float16*) alloc(FD * HD * 2);
  _Float16*  Wp2t    = (_Float16*) alloc(HD * HD * 2);
  _Float16*  Ws1t    = (_Float16*) alloc(2 * HD * HD * 2);
  _Float16*  Ws2t    = (_Float16*) alloc(2 * HD * HD * 2);
  float*     terms   = (float*)    alloc((size_t)ED * 4);
  (void)ws_size; (void)n_in; (void)in_sizes; (void)out_size;

  const int TM  = N / 16;        // 384 row tiles
  const int TNH = HD / 16;       // 8 col tiles (H=128)
  const int TNN = N / 16;        // 384 col tiles (sim)

  // ---- normalization precompute ----
  rowsum_rsqrt<<<N, 256, 0, stream>>>(adj, d1r, N);
  colsum_rsqrt<<<(N + 255) / 256, 256, 0, stream>>>(adj, d2r, N);
  prep_x<<<N, 128, 0, stream>>>(x, x_h, xn_h, cat1);
  f32_to_f16_t<<<(FD * HD + 255) / 256, 256, 0, stream>>>(Wg1, Wg1t, FD, HD);
  f32_to_f16_t<<<(HD * HD + 255) / 256, 256, 0, stream>>>(Wg2, Wg2t, HD, HD);
  f32_to_f16_t<<<(FD * HD + 255) / 256, 256, 0, stream>>>(Wp1, Wp1t, FD, HD);
  f32_to_f16_t<<<(HD * HD + 255) / 256, 256, 0, stream>>>(Wp2, Wp2t, HD, HD);
  f32_to_f16_t<<<(2 * HD * HD + 255) / 256, 256, 0, stream>>>(Ws1, Ws1t, 2 * HD, HD);
  f32_to_f16_t<<<(2 * HD * HD + 255) / 256, 256, 0, stream>>>(Ws2, Ws2t, 2 * HD, HD);
  make_anorm<<<(N / 256) * N, 256, 0, stream>>>(adj, d1r, d2r, anorm, N);

  // ---- GCN branch: h1 = elu(A @ (elu(A @ (x@Wg1)) @ Wg2)) ----
  wmma_gemm<false, false, true><<<TM * TNH / 8, 256, 0, stream>>>(
      x_h, FD, Wg1t, FD, nullptr, nullptr, 0, tT, N, TNH, FD);   // t1^T [128,N]
  gcn_agg<<<TM, 256, 0, stream>>>(anorm, tT, nullptr, 0, z1h, HD, N);  // z1 = elu(A@t1)
  wmma_gemm<false, false, true><<<TM * TNH / 8, 256, 0, stream>>>(
      z1h, HD, Wg2t, HD, nullptr, nullptr, 0, tT, N, TNH, HD);   // t2^T [128,N]
  gcn_agg<<<TM, 256, 0, stream>>>(anorm, tT, hbuf, 2 * HD, nullptr, 0, N); // h1 -> h[:,0:128]

  // ---- cosine-similarity graph + top-k ----
  wmma_gemm<false, false, false><<<TM * TNN / 8, 256, 0, stream>>>(
      xn_h, FD, xn_h, FD, nullptr, sim, N, nullptr, 0, TNN, FD); // sim = xn@xn^T
  topk_kernel<<<N, 256, 0, stream>>>(sim, ids, vals, deginv, N);

  // ---- SAGE layer 1 ----
  wmma_gemm<true, true, false><<<TM * TNH / 8, 256, 0, stream>>>(
      x_h, FD, Wp1t, FD, b1, hp, HD, nullptr, 0, TNH, FD);       // hp1 = elu(x@Wp1+b1)
  agg_kernel<<<N, 128, 0, stream>>>(hp, ids, vals, deginv, cat1, 2 * HD, HD);
  wmma_gemm<true, false, false><<<TM * TNH / 8, 256, 0, stream>>>(
      cat1, 2 * HD, Ws1t, 2 * HD, nullptr, nullptr, 0, cat2, 2 * HD, TNH, 2 * HD);

  // ---- SAGE layer 2 ----
  wmma_gemm<true, true, false><<<TM * TNH / 8, 256, 0, stream>>>(
      cat2, 2 * HD, Wp2t, HD, b2, hp, HD, nullptr, 0, TNH, HD);  // hp2 = elu(z2@Wp2+b2)
  agg_kernel<<<N, 128, 0, stream>>>(hp, ids, vals, deginv, cat2, 2 * HD, HD);
  wmma_gemm<true, false, false><<<TM * TNH / 8, 256, 0, stream>>>(
      cat2, 2 * HD, Ws2t, 2 * HD, nullptr, hbuf + HD, 2 * HD, nullptr, 0, TNH, 2 * HD);

  // ---- edge scoring + BCE loss ----
  edge_kernel<<<ED, 256, 0, stream>>>(src, dst, labels, hbuf, PredW, PredB, out, terms);
  loss_kernel<<<1, 256, 0, stream>>>(terms, out);
}
